// Attention_33741263077599
// MI455X (gfx1250) — compile-verified, matrix operand order checked by reading, not run on hardware
//
#include <hip/hip_runtime.h>
#include <hip/hip_bf16.h>
#include <math.h>

// ---------------- CDNA5 WMMA types / helpers ----------------
typedef __attribute__((ext_vector_type(16))) _Float16 v16h;
typedef __attribute__((ext_vector_type(8)))  float    v8f;
typedef unsigned int u32x4 __attribute__((ext_vector_type(4)));
typedef int          i32x4 __attribute__((ext_vector_type(4)));
typedef int          i32x8 __attribute__((ext_vector_type(8)));

#define D_MODEL 1024
#define NTOK    1025
#define NBATCH  8
#define NHEADS  16
#define DH      64
#define MROWS   (NBATCH * NTOK)   /* 8200 */
#define VTP     1152              /* padded token dim for V^T (>= 1088, mult of 16) */

union FragU { u32x4 u[2]; v16h v; };

// Load a 16-half WMMA operand fragment per the documented 16-bit A/B layout:
// lane half (hi = lane>>4) selects K-groups {0..7,16..23} or {8..15,24..31}.
__device__ __forceinline__ v16h load_frag(const _Float16* p, int base) {
  FragU f;
  f.u[0] = *(const u32x4*)(p + base);
  f.u[1] = *(const u32x4*)(p + 16 + base);
  return f.v;
}

__device__ __forceinline__ v8f wmma_f16(v16h a, v16h b, v8f c) {
  return __builtin_amdgcn_wmma_f32_16x16x32_f16(false, a, false, b, (short)0, c,
                                                false, false);
}

// ---------------- TDM: 2-D tiled async load Global -> LDS ----------------
// Builds a Tensor DMA Descriptor (D#) per CDNA5 ISA and
// issues TENSOR_LOAD_TO_LDS. data_size=2B. LDS padding: 4 DWORDs every
// 32 DWORDs -> effective LDS row stride 72 halves (bank-conflict-free and
// identical to the layout load_frag expects). OOB rows (y >= tensor_d1) are
// zero-filled by the TDM, which handles the M-tail for free.
__device__ __forceinline__ void tdm_load_2d(unsigned long long gaddr, // byte addr of tile origin
                                            unsigned lds_addr,        // LDS byte offset
                                            unsigned tensor_d0,       // elems valid along dim0
                                            unsigned tensor_d1,       // rows valid along dim1
                                            unsigned tile_d0,         // tile elems (k)
                                            unsigned tile_d1,         // tile rows
                                            unsigned long long stride0) { // row stride (elems)
  u32x4 g0;
  g0[0] = 1u;                                             // count=1, user descriptor
  g0[1] = lds_addr;                                       // LDS dest (bytes)
  g0[2] = (unsigned)(gaddr & 0xFFFFFFFFu);                // global addr lo
  g0[3] = (unsigned)((gaddr >> 32) & 0x01FFFFFFu) | (2u << 30); // addr hi | type=2
  i32x8 g1;
  //        data_size=2B   pad_en      pad_int=32dw  pad_amt=4dw
  g1[0] = (int)((1u << 16) | (1u << 20) | (4u << 22) | (3u << 25));
  g1[1] = (int)((tensor_d0 & 0xFFFFu) << 16);             // dim0 lo16 (bits 63:48)
  g1[2] = (int)(((tensor_d0 >> 16) & 0xFFFFu) | ((tensor_d1 & 0xFFFFu) << 16));
  g1[3] = (int)(((tensor_d1 >> 16) & 0xFFFFu) | ((tile_d0 & 0xFFFFu) << 16));
  g1[4] = (int)(tile_d1 & 0xFFFFu);                       // tile_dim1; tile_dim2=0
  g1[5] = (int)(unsigned)(stride0 & 0xFFFFFFFFu);         // dim0 stride lo32
  g1[6] = (int)(unsigned)((stride0 >> 32) & 0xFFFFu);     // stride hi16; dim1 stride=0
  g1[7] = 0;
  i32x4 z4 = {0, 0, 0, 0};
#if __clang_major__ >= 23
  i32x8 z8 = {0, 0, 0, 0, 0, 0, 0, 0};
  __builtin_amdgcn_tensor_load_to_lds(g0, g1, z4, z4, z8, 0);
#else
  __builtin_amdgcn_tensor_load_to_lds(g0, g1, z4, z4, 0);
#endif
}

// ---------------- elementwise kernels ----------------
__global__ void cvt_f16(const float* __restrict__ in, _Float16* __restrict__ out, int n) {
  int i = blockIdx.x * 256 + threadIdx.x;
  if (i < n) out[i] = (_Float16)in[i];
}

// wt[n*D + k] = w[k*D + n]  (store B transposed so GEMM B-fragments are contiguous)
__global__ void wtrans_f16(const float* __restrict__ w, _Float16* __restrict__ wt, int n) {
  int i = blockIdx.x * 256 + threadIdx.x;
  if (i >= n) return;
  int kc = i % D_MODEL;
  int nc = i / D_MODEL;
  wt[i] = (_Float16)w[(size_t)kc * D_MODEL + nc];
}

// In-place RoPE on f16 Q and K. Token 0 untouched; token t>=1 uses position t-1.
__global__ void rope_qk(_Float16* __restrict__ q, _Float16* __restrict__ k, int total) {
  int idx = blockIdx.x * 256 + threadIdx.x;
  if (idx >= total) return;
  int d  = idx & 31;
  int t1 = idx >> 5;
  int h  = t1 & 15;
  int t2 = t1 >> 4;
  int tok = t2 % NTOK;
  int b   = t2 / NTOK;
  if (tok == 0) return;
  float t   = (float)(tok - 1);
  float inv = powf(10000.0f, -((float)d) * (1.0f / 32.0f));
  float ang = t * inv;
  float s = sinf(ang), c = cosf(ang);
  size_t off = ((size_t)(b * NTOK + tok)) * D_MODEL + h * DH + d;
  float q1 = (float)q[off], q2 = (float)q[off + 32];
  q[off]      = (_Float16)(q1 * c - q2 * s);
  q[off + 32] = (_Float16)(q1 * s + q2 * c);
  float k1 = (float)k[off], k2 = (float)k[off + 32];
  k[off]      = (_Float16)(k1 * c - k2 * s);
  k[off + 32] = (_Float16)(k1 * s + k2 * c);
}

// vt[((b*H+h)*64 + d)*VTP + tok] = v[b,tok,h,d]; zero-fill padded tokens.
__global__ void vtrans_kernel(const _Float16* __restrict__ vh, _Float16* __restrict__ vt, int total) {
  int idx = blockIdx.x * 256 + threadIdx.x;
  if (idx >= total) return;
  int tok  = idx % VTP;
  int rest = idx / VTP;
  int d  = rest & 63;
  int bh = rest >> 6;
  int b = bh >> 4, h = bh & 15;
  _Float16 val = (_Float16)0.0f;
  if (tok < NTOK)
    val = vh[((size_t)(b * NTOK + tok)) * D_MODEL + h * DH + d];
  vt[idx] = val;
}

// ---------------- WMMA GEMM with TDM double-buffered staging ----------------
// C = A(MxK) * Bt^T + bias. A row-major f16; Bt row-major [N x K] f16.
// Block tile 128x64, BK=64, 8 waves in a 4x2 grid, each computing 32x32.
// Wave 0 drives the Tensor Data Mover: stage s+1 tiles stream into the
// alternate LDS buffer while all waves run WMMAs on stage s.
template <bool F16OUT>
__global__ void __launch_bounds__(256)
gemm_wmma(const _Float16* __restrict__ A, const _Float16* __restrict__ Bt,
          const float* __restrict__ bias, void* __restrict__ out,
          int M, int N, int K) {
  __shared__ __align__(16) _Float16 As[2][128 * 72];
  __shared__ __align__(16) _Float16 Bs[2][64 * 72];

  const int tid = threadIdx.x;
  const int lane = tid & 31, wave = tid >> 5;
  const int lane15 = lane & 15, hi = lane >> 4, base = hi * 8;
  const int wm = wave >> 1, wn = wave & 1;   // 4 x 2 wave grid, 32x32 per wave
  const int bM = blockIdx.x * 128, bN = blockIdx.y * 64;
  const int nstages = K >> 6;

  const unsigned long long gA0 = (unsigned long long)(uintptr_t)(A  + (size_t)bM * K);
  const unsigned long long gB0 = (unsigned long long)(uintptr_t)(Bt + (size_t)bN * K);
  const unsigned rowsA = (unsigned)(M - bM);             // TDM zero-fills rows >= rowsA
  const unsigned rowsB = (unsigned)(N - bN);
  const unsigned ldsA[2] = { (unsigned)(uintptr_t)&As[0][0], (unsigned)(uintptr_t)&As[1][0] };
  const unsigned ldsB[2] = { (unsigned)(uintptr_t)&Bs[0][0], (unsigned)(uintptr_t)&Bs[1][0] };

  v8f acc[2][2] = {};

  if (wave == 0) {   // prologue: stage 0 -> buffer 0
    tdm_load_2d(gA0, ldsA[0], 64u, rowsA, 64u, 128u, (unsigned long long)K);
    tdm_load_2d(gB0, ldsB[0], 64u, rowsB, 64u, 64u,  (unsigned long long)K);
  }

  for (int stage = 0; stage < nstages; ++stage) {
    const int cur = stage & 1;
    if (wave == 0) {
      if (stage + 1 < nstages) {   // stream next stage into the other buffer
        const unsigned long long koff = (unsigned long long)(stage + 1) * 64ull * 2ull;
        tdm_load_2d(gA0 + koff, ldsA[cur ^ 1], 64u, rowsA, 64u, 128u, (unsigned long long)K);
        tdm_load_2d(gB0 + koff, ldsB[cur ^ 1], 64u, rowsB, 64u, 64u,  (unsigned long long)K);
        __builtin_amdgcn_s_wait_tensorcnt(2);   // current pair done; next pair in flight
      } else {
        __builtin_amdgcn_s_wait_tensorcnt(0);
      }
    }
    __syncthreads();   // publish TDM-written LDS to all waves

    const _Float16* Ac = &As[cur][0];
    const _Float16* Bc = &Bs[cur][0];
#pragma unroll
    for (int ks = 0; ks < 2; ++ks) {
      const int kk = ks * 32;
      v16h af[2], bf[2];
#pragma unroll
      for (int mi = 0; mi < 2; ++mi)
        af[mi] = load_frag(Ac + (wm * 32 + mi * 16 + lane15) * 72 + kk, base);
#pragma unroll
      for (int nj = 0; nj < 2; ++nj)
        bf[nj] = load_frag(Bc + (wn * 32 + nj * 16 + lane15) * 72 + kk, base);
#pragma unroll
      for (int mi = 0; mi < 2; ++mi)
#pragma unroll
        for (int nj = 0; nj < 2; ++nj)
          acc[mi][nj] = wmma_f16(af[mi], bf[nj], acc[mi][nj]);
    }
    __syncthreads();   // all reads of buf[cur] done before it is refilled
  }

#pragma unroll
  for (int mi = 0; mi < 2; ++mi) {
#pragma unroll
    for (int nj = 0; nj < 2; ++nj) {
      int gc = bN + wn * 32 + nj * 16 + lane15;
      float bv = bias ? bias[gc] : 0.0f;
      int gr0 = bM + wm * 32 + mi * 16 + hi * 8;
#pragma unroll
      for (int v = 0; v < 8; ++v) {
        int gr = gr0 + v;
        if (gr < M) {
          float val = acc[mi][nj][v] + bv;
          if (F16OUT) ((_Float16*)out)[(size_t)gr * N + gc] = (_Float16)val;
          else        ((float*)out)[(size_t)gr * N + gc]    = val;
        }
      }
    }
  }
}

// ---------------- flash attention (per b,h, 64-query tile) ----------------
__global__ void __launch_bounds__(128)
attn_wmma(const _Float16* __restrict__ qh, const _Float16* __restrict__ kh,
          const _Float16* __restrict__ vt, _Float16* __restrict__ oh) {
  __shared__ __align__(16) _Float16 Qs[64 * 72];
  __shared__ __align__(16) _Float16 Ks[64 * 72];
  __shared__ __align__(16) _Float16 Vs[64 * 72];   // V^T tile: [d][key]
  __shared__ __align__(16) float    Ss[64 * 72];
  __shared__ __align__(16) _Float16 Ps[64 * 72];
  __shared__ float m_s[64], l_s[64], scl[64];

  const int tid = threadIdx.x;
  const int lane = tid & 31, wave = tid >> 5;
  const int lane15 = lane & 15, hi = lane >> 4, base = hi * 8;
  const int q0 = blockIdx.x * 64;
  const int bh = blockIdx.y;
  const int b = bh >> 4, h = bh & 15;

  for (int c = tid; c < 512; c += 128) {
    int r = c >> 3, dc = (c & 7) << 3;
    int tok = q0 + r;
    u32x4 v = {};
    if (tok < NTOK)
      v = *(const u32x4*)(qh + ((size_t)(b * NTOK + tok)) * D_MODEL + h * DH + dc);
    *(u32x4*)(Qs + r * 72 + dc) = v;
  }
  if (tid < 64) { m_s[tid] = -1e30f; l_s[tid] = 0.0f; }

  v8f accO[4] = {};

  for (int kt = 0; kt < 17; ++kt) {
    const int k0 = kt * 64;
    __syncthreads();
    for (int c = tid; c < 512; c += 128) {
      int r = c >> 3, dc = (c & 7) << 3;
      int tok = k0 + r;
      u32x4 vk = {};
      if (tok < NTOK)
        vk = *(const u32x4*)(kh + ((size_t)(b * NTOK + tok)) * D_MODEL + h * DH + dc);
      *(u32x4*)(Ks + r * 72 + dc) = vk;
      u32x4 vv = *(const u32x4*)(vt + ((size_t)bh * DH + r) * VTP + k0 + dc);
      *(u32x4*)(Vs + r * 72 + dc) = vv;
    }
    __syncthreads();

    v8f accS[4] = {};
#pragma unroll
    for (int ks = 0; ks < 2; ++ks) {
      v16h qa = load_frag(Qs + (wave * 16 + lane15) * 72 + ks * 32, base);
#pragma unroll
      for (int nj = 0; nj < 4; ++nj) {
        v16h kb = load_frag(Ks + (nj * 16 + lane15) * 72 + ks * 32, base);
        accS[nj] = wmma_f16(qa, kb, accS[nj]);
      }
    }
#pragma unroll
    for (int nj = 0; nj < 4; ++nj)
#pragma unroll
      for (int v = 0; v < 8; ++v)
        Ss[(wave * 16 + hi * 8 + v) * 72 + nj * 16 + lane15] = accS[nj][v];
    __syncthreads();

    if (tid < 64) {
      const int row = tid;
      const int kval = min(64, NTOK - k0);
      float mold = m_s[row];
      float mx = mold;
      for (int kk = 0; kk < kval; ++kk)
        mx = fmaxf(mx, Ss[row * 72 + kk] * 0.125f);
      float scale = __expf(mold - mx);
      float sum = 0.0f;
      for (int kk = 0; kk < 64; ++kk) {
        float p = (kk < kval) ? __expf(Ss[row * 72 + kk] * 0.125f - mx) : 0.0f;
        Ps[row * 72 + kk] = (_Float16)p;
        sum += p;
      }
      l_s[row] = l_s[row] * scale + sum;
      m_s[row] = mx;
      scl[row] = scale;
    }
    __syncthreads();

#pragma unroll
    for (int nj = 0; nj < 4; ++nj)
#pragma unroll
      for (int v = 0; v < 8; ++v)
        accO[nj][v] *= scl[wave * 16 + hi * 8 + v];
#pragma unroll
    for (int ks = 0; ks < 2; ++ks) {
      v16h pa = load_frag(Ps + (wave * 16 + lane15) * 72 + ks * 32, base);
#pragma unroll
      for (int nj = 0; nj < 4; ++nj) {
        v16h vb = load_frag(Vs + (nj * 16 + lane15) * 72 + ks * 32, base);
        accO[nj] = wmma_f16(pa, vb, accO[nj]);
      }
    }
  }

#pragma unroll
  for (int nj = 0; nj < 4; ++nj) {
#pragma unroll
    for (int v = 0; v < 8; ++v) {
      int row = wave * 16 + hi * 8 + v;
      int tok = q0 + row;
      if (tok < NTOK) {
        float o = accO[nj][v] / l_s[row];
        oh[((size_t)(b * NTOK + tok)) * D_MODEL + h * DH + nj * 16 + lane15] = (_Float16)o;
      }
    }
  }
}

// ---------------- launcher ----------------
extern "C" void kernel_launch(void* const* d_in, const int* in_sizes, int n_in,
                              void* d_out, int out_size, void* d_ws, size_t ws_size,
                              hipStream_t stream) {
  (void)in_sizes; (void)n_in; (void)out_size; (void)ws_size;
  const float* x  = (const float*)d_in[0];
  const float* wq = (const float*)d_in[1];
  const float* wk = (const float*)d_in[2];
  const float* wv = (const float*)d_in[3];
  const float* qb = (const float*)d_in[4];
  const float* vb = (const float*)d_in[5];
  const float* wo = (const float*)d_in[6];
  const float* bo = (const float*)d_in[7];

  char* ws = (char*)d_ws;
  size_t off = 0;
  auto alloc = [&](size_t elems) -> _Float16* {
    _Float16* p = (_Float16*)(ws + off);
    off += (elems * sizeof(_Float16) + 255) & ~(size_t)255;
    return p;
  };
  const size_t NX = (size_t)MROWS * D_MODEL;
  const size_t NW = (size_t)D_MODEL * D_MODEL;
  _Float16* xh  = alloc(NX);
  _Float16* wqt = alloc(NW);
  _Float16* wkt = alloc(NW);
  _Float16* wvt = alloc(NW);
  _Float16* wot = alloc(NW);
  _Float16* qh  = alloc(NX);
  _Float16* kh  = alloc(NX);
  _Float16* vh  = alloc(NX);
  _Float16* vtr = alloc((size_t)NBATCH * NHEADS * DH * VTP);
  _Float16* oh  = alloc(NX);

  cvt_f16<<<(int)((NX + 255) / 256), 256, 0, stream>>>(x, xh, (int)NX);
  wtrans_f16<<<(int)((NW + 255) / 256), 256, 0, stream>>>(wq, wqt, (int)NW);
  wtrans_f16<<<(int)((NW + 255) / 256), 256, 0, stream>>>(wk, wkt, (int)NW);
  wtrans_f16<<<(int)((NW + 255) / 256), 256, 0, stream>>>(wv, wvt, (int)NW);
  wtrans_f16<<<(int)((NW + 255) / 256), 256, 0, stream>>>(wo, wot, (int)NW);

  dim3 gg((MROWS + 127) / 128, D_MODEL / 64);
  gemm_wmma<true><<<gg, 256, 0, stream>>>(xh, wqt, qb, qh, MROWS, D_MODEL, D_MODEL);
  gemm_wmma<true><<<gg, 256, 0, stream>>>(xh, wkt, nullptr, kh, MROWS, D_MODEL, D_MODEL);
  gemm_wmma<true><<<gg, 256, 0, stream>>>(xh, wvt, vb, vh, MROWS, D_MODEL, D_MODEL);

  int nrope = NBATCH * NTOK * NHEADS * 32;
  rope_qk<<<(nrope + 255) / 256, 256, 0, stream>>>(qh, kh, nrope);

  int nvt = NBATCH * NHEADS * DH * VTP;
  vtrans_kernel<<<(nvt + 255) / 256, 256, 0, stream>>>(vh, vtr, nvt);

  dim3 ga((NTOK + 63) / 64, NBATCH * NHEADS);
  attn_wmma<<<ga, 128, 0, stream>>>(qh, kh, vtr, oh);

  gemm_wmma<false><<<gg, 256, 0, stream>>>(oh, wot, bo, d_out, MROWS, D_MODEL, D_MODEL);
}
